// ArchNetResNet_57818849739281
// MI455X (gfx1250) — matrix operand order repaired and checked by reading, not compiled, not run on hardware
//
#include <hip/hip_runtime.h>
#include <cstdint>
#include <cstddef>

// ---------------------------------------------------------------------------
// CDNA5 / gfx1250 WMMA types
// ---------------------------------------------------------------------------
typedef __attribute__((ext_vector_type(16))) _Float16 v16h;
typedef __attribute__((ext_vector_type(8)))  _Float16 v8h;
typedef __attribute__((ext_vector_type(8)))  float    v8f;
typedef __attribute__((ext_vector_type(4)))  int      v4i;

// Async global->LDS DMA (ASYNCcnt-tracked) if the toolchain exposes it.
#if defined(__has_builtin)
#  if __has_builtin(__builtin_amdgcn_global_load_async_to_lds_b128) && \
      __has_builtin(__builtin_amdgcn_s_wait_asynccnt)
#    define USE_ASYNC_LDS 1
#  endif
#endif
#ifndef USE_ASYNC_LDS
#  define USE_ASYNC_LDS 0
#endif

// ===========================================================================
// Phase A kernels: weight quantization (DoReFa) + BN folding
// ===========================================================================

__global__ void zero_f32_kernel(float* p, int n) {
  int i = blockIdx.x * blockDim.x + threadIdx.x;
  if (i < n) p[i] = 0.f;
}

// max over |tanh(w)| -> slot (float bits via atomicMax on uint; all values >= 0)
__global__ void wmax_kernel(const float* __restrict__ W, int n, float* slot) {
  __shared__ float red[256];
  float mx = 0.f;
  for (int i = blockIdx.x * blockDim.x + threadIdx.x; i < n;
       i += gridDim.x * blockDim.x)
    mx = fmaxf(mx, fabsf(tanhf(W[i])));
  red[threadIdx.x] = mx;
  __syncthreads();
  for (int s = 128; s > 0; s >>= 1) {
    if (threadIdx.x < s) red[threadIdx.x] = fmaxf(red[threadIdx.x], red[threadIdx.x + s]);
    __syncthreads();
  }
  if (threadIdx.x == 0) atomicMax((unsigned int*)slot, __float_as_uint(red[0]));
}

__device__ __forceinline__ float dorefa_w(float w, float M) {
  float t  = tanhf(w);
  float wn = t / (2.f * M) + 0.5f;
  float q  = rintf(wn * 255.f) * (1.f / 255.f);
  return 2.f * q - 1.f;
}

// Quantize conv weight [O][C][KH][KW] (f32) -> Wq[k][o] f16, k = (c*KH+kh)*KW+kw,
// rows k in [K, Kpad) zero-filled (K padded to multiple of 32 for BK=32 loop).
__global__ void wquant_conv_kernel(const float* __restrict__ W,
                                   const float* __restrict__ slot,
                                   _Float16* __restrict__ Wq,
                                   int O, int C, int KH, int KW, int K, int Kpad) {
  int idx = blockIdx.x * blockDim.x + threadIdx.x;
  int total = Kpad * O;
  if (idx >= total) return;
  int k = idx / O, o = idx - k * O;
  float out = 0.f;
  if (k < K) {
    int khw = KH * KW;
    int c = k / khw;
    int r = k - c * khw;
    int kh = r / KW, kw = r - kh * KW;
    out = dorefa_w(W[((size_t)(o * C + c) * KH + kh) * KW + kw], slot[0]);
  }
  Wq[idx] = (_Float16)out;
}

// FC weight: keep original [O][K] layout, elementwise quantize.
__global__ void wquant_flat_kernel(const float* __restrict__ W,
                                   const float* __restrict__ slot,
                                   _Float16* __restrict__ Wq, int n) {
  int i = blockIdx.x * blockDim.x + threadIdx.x;
  if (i >= n) return;
  Wq[i] = (_Float16)dorefa_w(W[i], slot[0]);
}

// Fold BN (inference) into per-channel scale/bias: inv = g/sqrt(v+eps), bias = b - m*inv
__global__ void bnprep_kernel(const float* __restrict__ g, const float* __restrict__ b,
                              const float* __restrict__ m, const float* __restrict__ v,
                              int n, float* __restrict__ inv, float* __restrict__ bias) {
  int i = blockIdx.x * blockDim.x + threadIdx.x;
  if (i >= n) return;
  float iv = g[i] * rsqrtf(v[i] + 1e-5f);
  inv[i]  = iv;
  bias[i] = b[i] - m[i] * iv;
}

// ===========================================================================
// Forward kernels
// ===========================================================================

__global__ void cast_f32_f16_kernel(const float* __restrict__ x,
                                    _Float16* __restrict__ y, int n) {
  int i = blockIdx.x * blockDim.x + threadIdx.x;
  if (i < n) y[i] = (_Float16)x[i];
}

// ---------------------------------------------------------------------------
// Implicit-GEMM conv + fused BN (+ optional ReLU + DoReFa activation quant).
//   GEMM:  C[M=B*Ho*Wo][N=O] = A[M][K] * B[K][N],  K = Cin*KH*KW (padded)
//   Tile:  64 x 64, BK = 32, double-buffered LDS.  256 threads = 8 wave32,
//          wave grid 4(M) x 2(N): each wave = one 16x32 A fragment and two
//          v_wmma_f32_16x16x32_f16 accumulations per K step.
//   Weight tile is DMA'd with GLOBAL_LOAD_ASYNC_TO_LDS_B128 (ASYNCcnt) when
//   available; the next tile's staging is issued before the current WMMAs so
//   global traffic overlaps matrix math.
// VGPR fragment layouts per CDNA5 ISA 7.12.2:
//   A 16x32 f16 : lane(M%16), half=lane>>4; v0..3 -> K = half*8 + {0..7},
//                 v4..7 -> K = 16 + half*8 + {0..7}  (f16 pairs per VGPR)
//   B 32x16 f16 : lane -> N = lane&15, half=lane>>4; K order same as A
//   C/D 16x16   : lane -> N = lane&15, VGPR p -> M = p + 8*(lane>>4)
// ---------------------------------------------------------------------------
template <int KH, int KW, int STRIDE, int PAD, bool RELUQ>
__global__ __launch_bounds__(256) void conv_wmma_kernel(
    const _Float16* __restrict__ X,   // [B][cinTot][Hin][Win] f16 (channel window)
    const _Float16* __restrict__ Wq,  // [Kpad][O] f16 (DoReFa-quantized)
    const float* __restrict__ bninv, const float* __restrict__ bnbias,
    const float* __restrict__ qscale,
    _Float16* __restrict__ Y,         // [B][coutTot][Hout][Wout] f16
    int Bn, int Hin, int Win, int O, int Hout, int Wout,
    int K, int Kpad, int cinTot, int cinOff, int coutTot, int coutOff) {
  __shared__ __align__(16) _Float16 As[2][64][40];  // 64 M x 32 K (+pad), x2
  __shared__ __align__(16) _Float16 Bs[2][32][72];  // 32 K x 64 N (+pad), x2

  const int tid = threadIdx.x;
  const int HWo = Hout * Wout;
  const int M   = Bn * HWo;
  const int m0  = blockIdx.x * 64;
  const int n0  = blockIdx.y * 64;

  // A cooperative-load mapping: thread -> (m_local = tid&63, k rows kg*8..+7)
  const int ml = tid & 63;
  const int kg = tid >> 6;  // 0..3
  const int gm = m0 + ml;
  const bool mvalid = (gm < M);
  int bImg = 0, oh = 0, ow = 0;
  if (mvalid) {
    bImg = gm / HWo;
    int pix = gm - bImg * HWo;
    oh = pix / Wout;
    ow = pix - oh * Wout;
  }
  // B cooperative-load mapping: thread -> (k = tid>>3, 8 contiguous n = 16B)
  const int kb = tid >> 3;        // 0..31
  const int nb = (tid & 7) * 8;   // 0..56
  const bool fullN = (n0 + 64 <= O);  // uniform: whole tile in-range -> DMA path

  // wave/fragment mapping
  const int lane  = tid & 31;
  const int wv    = tid >> 5;   // 0..7 (wave32)
  const int wm    = wv & 3;     // M tile 0..3
  const int wn    = wv >> 2;    // N tile 0..1
  const int halfw = lane >> 4;
  const int rowA  = wm * 16 + (lane & 15);

  // ---- tile staging helpers ----
  auto stageA = [&](int buf, int k0) {
    v8h va;
#pragma unroll
    for (int j = 0; j < 8; ++j) {
      int kk = k0 + kg * 8 + j;
      _Float16 val = (_Float16)0.f;
      if (mvalid && kk < K) {
        int c, kh, kw;
        if (KH == 1 && KW == 1) {
          c = kk; kh = 0; kw = 0;
        } else {
          c = kk / (KH * KW);
          int r = kk - c * (KH * KW);
          kh = r / KW;
          kw = r - kh * KW;
        }
        int hi = oh * STRIDE - PAD + kh;
        int wi = ow * STRIDE - PAD + kw;
        if (hi >= 0 && hi < Hin && wi >= 0 && wi < Win)
          val = X[(((size_t)bImg * cinTot + cinOff + c) * Hin + hi) * Win + wi];
      }
      va[j] = val;
    }
    *(v8h*)&As[buf][ml][kg * 8] = va;  // 16B ds_store
  };

  auto stageB = [&](int buf, int k0) {
    const _Float16* src = &Wq[(size_t)(k0 + kb) * O + n0 + nb];
#if USE_ASYNC_LDS
    if (fullN) {
      // async 16B/lane DMA: global -> LDS, tracked by ASYNCcnt
      __builtin_amdgcn_global_load_async_to_lds_b128(
          (__attribute__((address_space(1))) v4i*)src,
          (__attribute__((address_space(3))) v4i*)&Bs[buf][kb][nb], 0, 0);
      return;
    }
#endif
    if (fullN) {
      *(v8h*)&Bs[buf][kb][nb] = *(const v8h*)src;  // 16B load/store
#if !USE_ASYNC_LDS
      if (k0 + 32 < Kpad)  // global_prefetch_b8 next weight tile
        __builtin_prefetch(src + (size_t)32 * O, 0, 1);
#endif
    } else {
      v8h vb;
#pragma unroll
      for (int j = 0; j < 8; ++j)
        vb[j] = (n0 + nb + j < O) ? src[j] : (_Float16)0.f;
      *(v8h*)&Bs[buf][kb][nb] = vb;
    }
  };

  v8f acc0 = {0.f, 0.f, 0.f, 0.f, 0.f, 0.f, 0.f, 0.f};
  v8f acc1 = {0.f, 0.f, 0.f, 0.f, 0.f, 0.f, 0.f, 0.f};

  // ---- prologue: stage tile 0 ----
  stageA(0, 0);
  stageB(0, 0);
#if USE_ASYNC_LDS
  if (fullN) __builtin_amdgcn_s_wait_asynccnt(0);
#endif
  __syncthreads();

  int cur = 0;
  for (int k0 = 0; k0 < Kpad; k0 += 32) {
    // ---- pull fragments into VGPRs (16B ds_load_b128 each) ----
    v8h alo = *(const v8h*)&As[cur][rowA][halfw * 8];
    v8h ahi = *(const v8h*)&As[cur][rowA][16 + halfw * 8];
    v16h a = __builtin_shufflevector(alo, ahi, 0, 1, 2, 3, 4, 5, 6, 7, 8, 9, 10,
                                     11, 12, 13, 14, 15);
    v16h b0, b1;
#pragma unroll
    for (int s = 0; s < 16; ++s) {
      int kr = (s < 8) ? (halfw * 8 + s) : (16 + halfw * 8 + (s - 8));
      b0[s] = Bs[cur][kr][wn * 32 + (lane & 15)];
      b1[s] = Bs[cur][kr][wn * 32 + 16 + (lane & 15)];
    }

    // ---- overlap: issue next tile's staging before the matrix ops ----
    const bool hasNext = (k0 + 32 < Kpad);  // uniform
    if (hasNext) {
      stageA(cur ^ 1, k0 + 32);
      stageB(cur ^ 1, k0 + 32);
    }

    // ---- matrix math (EXEC all-ones here; staging predication has retired) ----
    acc0 = __builtin_amdgcn_wmma_f32_16x16x32_f16(false, a, false, b0,
                                                  (short)0, acc0, false, false);
    acc1 = __builtin_amdgcn_wmma_f32_16x16x32_f16(false, a, false, b1,
                                                  (short)0, acc1, false, false);

#if USE_ASYNC_LDS
    if (fullN && hasNext) __builtin_amdgcn_s_wait_asynccnt(0);
#endif
    __syncthreads();
    cur ^= 1;
  }

  // ---- epilogue: BN (+ ReLU + k-bit activation quant), NCHW f16 store ----
#pragma unroll
  for (int nt = 0; nt < 2; ++nt) {
    v8f acc = (nt == 0) ? acc0 : acc1;
    int gn = n0 + wn * 32 + nt * 16 + (lane & 15);
    if (gn < O) {
      float inv = bninv[gn];
      float bia = bnbias[gn];
      float s   = RELUQ ? qscale[gn] : 1.f;
#pragma unroll
      for (int p = 0; p < 8; ++p) {
        int gmm = m0 + wm * 16 + halfw * 8 + p;
        if (gmm < M) {
          float z = acc[p] * inv + bia;
          if (RELUQ) {
            z = fmaxf(z, 0.f);
            float t = fminf(z / s, 1.f);
            z = s * rintf(t * 255.f) * (1.f / 255.f);
          }
          int bI  = gmm / HWo;
          int pix = gmm - bI * HWo;
          Y[((size_t)bI * coutTot + coutOff + gn) * HWo + pix] = (_Float16)z;
        }
      }
    }
  }
}

// 2x2/2 max pool, NCHW f16
__global__ void maxpool_kernel(const _Float16* __restrict__ X,
                               _Float16* __restrict__ Y, int BC, int H, int W) {
  int Ho = H >> 1, Wo = W >> 1;
  int total = BC * Ho * Wo;
  int i = blockIdx.x * blockDim.x + threadIdx.x;
  if (i >= total) return;
  int bc = i / (Ho * Wo);
  int p  = i - bc * Ho * Wo;
  int oh = p / Wo, ow = p - oh * Wo;
  const _Float16* s = X + ((size_t)bc * H + oh * 2) * W + ow * 2;
  float a = (float)s[0], b = (float)s[1], c = (float)s[W], d = (float)s[W + 1];
  Y[i] = (_Float16)fmaxf(fmaxf(a, b), fmaxf(c, d));
}

// residual: quant_act(relu(u + idn), s) per output channel
__global__ void addquant_kernel(const _Float16* __restrict__ U,
                                const _Float16* __restrict__ I,
                                const float* __restrict__ qs,
                                _Float16* __restrict__ Y, int total, int C, int HW) {
  int i = blockIdx.x * blockDim.x + threadIdx.x;
  if (i >= total) return;
  int c = (i / HW) % C;
  float z = (float)U[i] + (float)I[i];
  z = fmaxf(z, 0.f);
  float s = qs[c];
  float t = fminf(z / s, 1.f);
  Y[i] = (_Float16)(s * rintf(t * 255.f) * (1.f / 255.f));
}

// global average pool 7x7 -> [B*C] f32
__global__ void avgpool_kernel(const _Float16* __restrict__ X, float* __restrict__ Y,
                               int BC, int HW) {
  int i = blockIdx.x * blockDim.x + threadIdx.x;
  if (i >= BC) return;
  const _Float16* s = X + (size_t)i * HW;
  float acc = 0.f;
  for (int j = 0; j < HW; ++j) acc += (float)s[j];
  Y[i] = acc / (float)HW;
}

// FC + int16 logit quantizer (out_scale = 32, n = 2^15 - 1)
__global__ void fc_kernel(const float* __restrict__ h,       // [4][2048]
                          const _Float16* __restrict__ Wq,   // [1000][2048]
                          const float* __restrict__ bias,
                          float* __restrict__ out) {         // [4][1000]
  int i = blockIdx.x * blockDim.x + threadIdx.x;
  if (i >= 4 * 1000) return;
  int b = i / 1000, o = i - b * 1000;
  const float*    hp = h + (size_t)b * 2048;
  const _Float16* wp = Wq + (size_t)o * 2048;
  float acc = 0.f;
  for (int k = 0; k < 2048; ++k) acc += hp[k] * (float)wp[k];
  acc += bias[o];
  float t = fminf(fmaxf(acc * (1.f / 32.f), -1.f), 1.f);
  const float n = 32767.f;
  out[i] = 32.f * rintf(t * n) / n;
}

// ===========================================================================
// Host orchestration
// ===========================================================================

struct ConvW {
  const float *w, *g, *b, *m, *v, *s;
  int O, C, KH, KW, K, Kpad;
  _Float16* wq;
  float *inv, *bias, *slot;
};

struct BlkRec {
  int ca, cb, cc, cd1, cd2;
  const float* sOut;
};

static void launch_conv(hipStream_t st, const ConvW& d, const _Float16* X,
                        _Float16* Y, int Hin, int Win, int stride, bool reluq,
                        int cinTot, int cinOff, int coutTot, int coutOff) {
  const int Bn = 4;
  int pad  = (d.KH == 3) ? 1 : 0;
  int Hout = (Hin + 2 * pad - d.KH) / stride + 1;
  int Wout = (Win + 2 * pad - d.KW) / stride + 1;
  long M = (long)Bn * Hout * Wout;
  dim3 grid((unsigned)((M + 63) / 64), (unsigned)((d.O + 63) / 64), 1);
  dim3 blk(256, 1, 1);
#define CARGS X, d.wq, d.inv, d.bias, d.s, Y, Bn, Hin, Win, d.O, Hout, Wout, \
              d.K, d.Kpad, cinTot, cinOff, coutTot, coutOff
  if (d.KH == 1) {
    if (reluq) conv_wmma_kernel<1, 1, 1, 0, true><<<grid, blk, 0, st>>>(CARGS);
    else       conv_wmma_kernel<1, 1, 1, 0, false><<<grid, blk, 0, st>>>(CARGS);
  } else {
    if (stride == 1) conv_wmma_kernel<3, 3, 1, 1, true><<<grid, blk, 0, st>>>(CARGS);
    else             conv_wmma_kernel<3, 3, 2, 1, true><<<grid, blk, 0, st>>>(CARGS);
  }
#undef CARGS
}

extern "C" void kernel_launch(void* const* d_in, const int* in_sizes, int n_in,
                              void* d_out, int out_size, void* d_ws, size_t ws_size,
                              hipStream_t stream) {
  (void)in_sizes; (void)n_in; (void)out_size; (void)ws_size;

  // ---- walk params in setup_inputs() dict insertion order ----
  int cur = 0;
  auto nf = [&]() -> const float* { return (const float*)d_in[cur++]; };
  const float* x32 = nf();

  ConvW cw[64];
  int nc = 0;
  auto mk = [&](const float* w, const float* g, const float* b, const float* m,
                const float* v, const float* s, int O, int C, int KH, int KW) -> int {
    ConvW& d = cw[nc];
    d.w = w; d.g = g; d.b = b; d.m = m; d.v = v; d.s = s;
    d.O = O; d.C = C; d.KH = KH; d.KW = KW;
    d.K = C * KH * KW;
    d.Kpad = (d.K + 31) & ~31;
    return nc++;
  };
  auto mkseq = [&](int O, int C, int KH, int KW) -> int {
    const float* w = nf();
    const float* g = nf(); const float* b = nf();
    const float* m = nf(); const float* v = nf();
    const float* s = nf();
    return mk(w, g, b, m, v, s, O, C, KH, KW);
  };

  const int PL[4] = {64, 128, 256, 512};
  const int NB[4] = {3, 4, 6, 3};

  int stemI[3];
  int ic = 3;
  for (int i = 0; i < 3; ++i) { stemI[i] = mkseq(64, ic, 3, 3); ic = 64; }

  BlkRec br[16];
  int nbk = 0;
  int inp = 64;
  for (int li = 0; li < 4; ++li) {
    int pl = PL[li];
    for (int bi = 0; bi < NB[li]; ++bi) {
      const float* w0 = nf(); const float* w1 = nf(); const float* w2 = nf();
      const float* bnp[3][4];
      for (int j = 0; j < 3; ++j)
        for (int t = 0; t < 4; ++t) bnp[j][t] = nf();
      const float* s0 = nf(); const float* s1 = nf(); const float* s2 = nf();
      BlkRec& r = br[nbk++];
      r.ca = mk(w0, bnp[0][0], bnp[0][1], bnp[0][2], bnp[0][3], s0, pl, inp, 1, 1);
      r.cb = mk(w1, bnp[1][0], bnp[1][1], bnp[1][2], bnp[1][3], s1, pl, pl, 3, 3);
      r.cc = mk(w2, bnp[2][0], bnp[2][1], bnp[2][2], bnp[2][3], s2, pl * 4, pl, 1, 1);
      r.sOut = s2;
      r.cd1 = r.cd2 = -1;
      if (bi == 0) {
        if (li == 3) {             // SplitDownsample: two 512->1024 3x3/s2 convs
          r.cd1 = mkseq(1024, 512, 3, 3);
          r.cd2 = mkseq(1024, 512, 3, 3);
        } else {
          r.cd1 = mkseq(pl * 4, inp, 3, 3);
        }
      }
      inp = pl * 4;
    }
  }
  const float* fcw = nf();
  const float* fcb = nf();

  // ---- workspace bump allocator ----
  char* base = (char*)d_ws;
  size_t off = 0;
  auto alloc = [&](size_t bytes) -> void* {
    off = (off + 255) & ~(size_t)255;
    void* r = base + off;
    off += bytes;
    return r;
  };
  float* slots = (float*)alloc((size_t)(nc + 1) * sizeof(float));
  for (int i = 0; i < nc; ++i) {
    cw[i].slot = slots + i;
    cw[i].wq   = (_Float16*)alloc((size_t)cw[i].Kpad * cw[i].O * sizeof(_Float16));
    cw[i].inv  = (float*)alloc((size_t)cw[i].O * sizeof(float));
    cw[i].bias = (float*)alloc((size_t)cw[i].O * sizeof(float));
  }
  float* fcslot = slots + nc;
  _Float16* fcwq = (_Float16*)alloc((size_t)1000 * 2048 * sizeof(_Float16));
  const size_t TSZ = (size_t)4 * 64 * 224 * 224;  // largest live tensor (elems)
  _Float16* T0 = (_Float16*)alloc(TSZ * sizeof(_Float16));
  _Float16* T1 = (_Float16*)alloc(TSZ * sizeof(_Float16));
  _Float16* T2 = (_Float16*)alloc(TSZ * sizeof(_Float16));
  float* pooled = (float*)alloc((size_t)4 * 2048 * sizeof(float));

  // ---- Phase A: weight max-abs(tanh), BN fold, quantize ----
  zero_f32_kernel<<<1, 256, 0, stream>>>(slots, nc + 1);
  for (int i = 0; i < nc; ++i) {
    int n = cw[i].O * cw[i].C * cw[i].KH * cw[i].KW;
    int g = (n + 255) / 256; if (g > 512) g = 512;
    wmax_kernel<<<g, 256, 0, stream>>>(cw[i].w, n, cw[i].slot);
  }
  {
    int n = 1000 * 2048;
    wmax_kernel<<<512, 256, 0, stream>>>(fcw, n, fcslot);
  }
  for (int i = 0; i < nc; ++i) {
    bnprep_kernel<<<(cw[i].O + 255) / 256, 256, 0, stream>>>(
        cw[i].g, cw[i].b, cw[i].m, cw[i].v, cw[i].O, cw[i].inv, cw[i].bias);
    int tot = cw[i].Kpad * cw[i].O;
    wquant_conv_kernel<<<(tot + 255) / 256, 256, 0, stream>>>(
        cw[i].w, cw[i].slot, cw[i].wq, cw[i].O, cw[i].C, cw[i].KH, cw[i].KW,
        cw[i].K, cw[i].Kpad);
  }
  wquant_flat_kernel<<<(1000 * 2048 + 255) / 256, 256, 0, stream>>>(
      fcw, fcslot, fcwq, 1000 * 2048);

  // ---- Phase B: forward ----
  {
    int n = 4 * 3 * 224 * 224;
    cast_f32_f16_kernel<<<(n + 255) / 256, 256, 0, stream>>>(x32, T0, n);
  }
  // stem: 3x3/s1 (3->64), 3x3/s1, 3x3/s2, then 2x2 maxpool
  launch_conv(stream, cw[stemI[0]], T0, T1, 224, 224, 1, true, 3, 0, 64, 0);
  launch_conv(stream, cw[stemI[1]], T1, T2, 224, 224, 1, true, 64, 0, 64, 0);
  launch_conv(stream, cw[stemI[2]], T2, T0, 224, 224, 2, true, 64, 0, 64, 0);
  {
    int BC = 4 * 64, H = 112, W = 112;
    int tot = BC * (H / 2) * (W / 2);
    maxpool_kernel<<<(tot + 255) / 256, 256, 0, stream>>>(T0, T1, BC, H, W);
  }

  _Float16 *X = T1, *Ta = T2, *Tb = T0;
  int Cin = 64, S = 56, bidx = 0;
  for (int li = 0; li < 4; ++li) {
    int pl = PL[li];
    for (int bi = 0; bi < NB[li]; ++bi) {
      int st = (bi == 0 && li > 0) ? 2 : 1;
      int So = S / st;
      const BlkRec& r = br[bidx++];
      launch_conv(stream, cw[r.ca], X, Ta, S, S, 1, true, Cin, 0, pl, 0);
      launch_conv(stream, cw[r.cb], Ta, Tb, S, S, st, true, pl, 0, pl, 0);
      launch_conv(stream, cw[r.cc], Tb, Ta, So, So, 1, false, pl, 0, pl * 4, 0);
      const _Float16* idn;
      if (r.cd2 >= 0) {  // split downsample, concat halves via channel offsets
        launch_conv(stream, cw[r.cd1], X, Tb, S, S, st, true, 1024, 0, 2048, 0);
        launch_conv(stream, cw[r.cd2], X, Tb, S, S, st, true, 1024, 512, 2048, 1024);
        idn = Tb;
      } else if (r.cd1 >= 0) {
        launch_conv(stream, cw[r.cd1], X, Tb, S, S, st, true, Cin, 0, pl * 4, 0);
        idn = Tb;
      } else {
        idn = X;
      }
      int C = pl * 4, HW = So * So, tot = 4 * C * HW;
      if (idn == X) {
        addquant_kernel<<<(tot + 255) / 256, 256, 0, stream>>>(Ta, idn, r.sOut,
                                                               Tb, tot, C, HW);
        _Float16* t = X; X = Tb; Tb = Ta; Ta = t;  // rotate buffers
      } else {
        addquant_kernel<<<(tot + 255) / 256, 256, 0, stream>>>(Ta, idn, r.sOut,
                                                               X, tot, C, HW);
      }
      Cin = C;
      S = So;
    }
  }

  // avgpool + FC (+ int16 logit quantizer) -> d_out f32 [4][1000]
  avgpool_kernel<<<(4 * 2048 + 255) / 256, 256, 0, stream>>>(X, pooled, 4 * 2048,
                                                             S * S);
  fc_kernel<<<(4000 + 255) / 256, 256, 0, stream>>>(pooled, fcwq, fcb,
                                                    (float*)d_out);
}
